// EvidentialHead_9328668966958
// MI455X (gfx1250) — compile-verified, matrix operand order checked by reading, not run on hardware
//
#include <hip/hip_runtime.h>
#include <hip/hip_bf16.h>

// ---------------------------------------------------------------------------
// EvidentialHead on MI455X (gfx1250)
//   - split-bf16 (3-term) WMMA GEMMs, near-fp32 accuracy at bf16-pipe rate
//   - weights pre-split + pre-transposed once -> bf16 B-path
//   - double-buffered LDS, async global->LDS copies, 1 barrier per K-step
// ---------------------------------------------------------------------------

typedef __attribute__((ext_vector_type(16))) __bf16 v16bf;
typedef __attribute__((ext_vector_type(8)))  __bf16 v8bf;
typedef __attribute__((ext_vector_type(4)))  __bf16 v4bf;
typedef __attribute__((ext_vector_type(8)))  float  v8f;
typedef __attribute__((ext_vector_type(4)))  int    v4i;

#define B_ROWS 8192
#define D_DIM  1024
#define H_DIM  1024
#define K_CLS  100

#if defined(__has_builtin)
#if __has_builtin(__builtin_amdgcn_global_load_async_to_lds_b128) && \
    __has_builtin(__builtin_amdgcn_s_wait_asynccnt)
#define USE_ASYNC_LDS 1
#endif
#endif
#ifndef USE_ASYNC_LDS
#define USE_ASYNC_LDS 0
#endif

#if USE_ASYNC_LDS
typedef v4i __attribute__((address_space(1)))* as1_v4ip;
typedef v4i __attribute__((address_space(3)))* as3_v4ip;
__device__ __forceinline__ as1_v4ip to_global(const void* p) {
    return (as1_v4ip)(unsigned long long)p;
}
__device__ __forceinline__ as3_v4ip to_lds(void* p) {
    // flat LDS address maps to LDS offset by low-32-bit truncation (ISA 10.2)
    return (as3_v4ip)(unsigned int)(unsigned long long)p;
}
#endif

// split f32 into bf16 hi + bf16 lo (a ~= hi + lo)
__device__ __forceinline__ void splitbf(float f, __bf16& h, __bf16& l) {
    h = (__bf16)f;
    l = (__bf16)(f - (float)h);
}

// Build a 16x32 (A) / 32x16 (B) bf16 WMMA fragment from an LDS row of 32
// contiguous bf16. Lanes 0-15: K 0-7 (v0-3), K 16-23 (v4-7); lanes 16-31:
// K 8-15, K 24-31.
__device__ __forceinline__ v16bf load_frag(const __bf16* p, int k0) {
    v8bf a = *(const v8bf*)(p + k0);
    v8bf b = *(const v8bf*)(p + k0 + 16);
    return __builtin_shufflevector(a, b, 0, 1, 2, 3, 4, 5, 6, 7,
                                         8, 9, 10, 11, 12, 13, 14, 15);
}

// ---------------------------------------------------------------------------
// One-time weight prep: f32 [R x C] -> transposed split bf16 hi/lo [C x R]
// ---------------------------------------------------------------------------
__global__ __launch_bounds__(256)
void transpose_split(const float* __restrict__ src, __bf16* __restrict__ dh,
                     __bf16* __restrict__ dl, int R, int C) {
    __shared__ float t[32][33];
    const int tx = threadIdx.x & 31, ty = threadIdx.x >> 5;   // 32 x 8
    const int bx = blockIdx.x, by = blockIdx.y;
#pragma unroll
    for (int i = 0; i < 4; ++i) {
        int r = by * 32 + ty + i * 8, c = bx * 32 + tx;
        t[ty + i * 8][tx] = (r < R && c < C) ? src[(size_t)r * C + c] : 0.0f;
    }
    __syncthreads();
#pragma unroll
    for (int i = 0; i < 4; ++i) {
        int oc  = bx * 32 + ty + i * 8;   // original col -> output row
        int orw = by * 32 + tx;           // original row -> output col
        if (oc < C && orw < R) {
            __bf16 h, l;
            splitbf(t[tx][ty + i * 8], h, l);
            dh[(size_t)oc * R + orw] = h;
            dl[(size_t)oc * R + orw] = l;
        }
    }
}

// ---------------------------------------------------------------------------
// GEMM: out[M x N] = act(A[M x K] @ B[K x N] + bias[N])
//  A: f32, split on the fly.  B: pre-split bf16, transposed (Bt = [N x K]).
//  block = 256 threads (8 waves), tile 128(M) x 64(N), K-step 32,
//  double-buffered LDS with one barrier per K-step.
//  ACT: 0 = identity, 1 = SiLU
// ---------------------------------------------------------------------------
template <int ACT>
__global__ __launch_bounds__(256)
void gemm_bias_act(const float* __restrict__ A,
                   const __bf16* __restrict__ Bt_h,
                   const __bf16* __restrict__ Bt_l,
                   const float* __restrict__ bias, float* __restrict__ out,
                   int M, int N, int K) {
    __shared__ __bf16 As_h[2][128][40];
    __shared__ __bf16 As_l[2][128][40];
    __shared__ __bf16 Bs_h[2][64][40];
    __shared__ __bf16 Bs_l[2][64][40];

    const int tid     = threadIdx.x;
    const int lane    = tid & 31;
    const int wave    = tid >> 5;
    const int lane16  = lane & 15;
    const int laneHi  = lane >> 4;
    const int k0      = laneHi * 8;
    const int waveM   = wave >> 1;           // 0..3
    const int waveN   = wave & 1;            // 0..1
    const int rowBase = blockIdx.x * 128;
    const int colBase = blockIdx.y * 64;

    // B staging assignment: one 16B chunk (8 bf16) per thread per matrix
    const int bn = tid >> 2;                 // 0..63   (tile row = N)
    const int bk = (tid & 3) * 8;            // 0,8,16,24
    const bool bInRange = (colBase + bn) < N;

    // zero-fill OOB B rows once (only the last N-block of GEMM3 takes this)
    if (!bInRange) {
        v8bf z = (v8bf)((__bf16)0.0f);
        *(v8bf*)&Bs_h[0][bn][bk] = z;  *(v8bf*)&Bs_l[0][bn][bk] = z;
        *(v8bf*)&Bs_h[1][bn][bk] = z;  *(v8bf*)&Bs_l[1][bn][bk] = z;
    }

    // stage helpers ---------------------------------------------------------
    auto issueB = [&](int kb, int buf) {
        if (bInRange) {
            const __bf16* gh = Bt_h + (size_t)(colBase + bn) * K + kb + bk;
            const __bf16* gl = Bt_l + (size_t)(colBase + bn) * K + kb + bk;
#if USE_ASYNC_LDS
            __builtin_amdgcn_global_load_async_to_lds_b128(
                to_global(gh), to_lds(&Bs_h[buf][bn][bk]), 0, 0);
            __builtin_amdgcn_global_load_async_to_lds_b128(
                to_global(gl), to_lds(&Bs_l[buf][bn][bk]), 0, 0);
#else
            *(v8bf*)&Bs_h[buf][bn][bk] = *(const v8bf*)gh;
            *(v8bf*)&Bs_l[buf][bn][bk] = *(const v8bf*)gl;
#endif
        }
    };
    auto loadA = [&](int kb, float4* areg) {
#pragma unroll
        for (int i = 0; i < 4; ++i) {
            int idx = tid + i * 256;               // 0..1023
            int row = idx >> 3;                    // 0..127
            int c4  = (idx & 7) * 4;               // 0..28
            areg[i] = *(const float4*)(A + (size_t)(rowBase + row) * K + kb + c4);
        }
    };
    auto storeA = [&](const float4* areg, int buf) {
#pragma unroll
        for (int i = 0; i < 4; ++i) {
            int idx = tid + i * 256;
            int row = idx >> 3;
            int c4  = (idx & 7) * 4;
            __bf16 h0, h1, h2, h3, l0, l1, l2, l3;
            splitbf(areg[i].x, h0, l0);
            splitbf(areg[i].y, h1, l1);
            splitbf(areg[i].z, h2, l2);
            splitbf(areg[i].w, h3, l3);
            v4bf hv = {h0, h1, h2, h3};
            v4bf lv = {l0, l1, l2, l3};
            *(v4bf*)&As_h[buf][row][c4] = hv;
            *(v4bf*)&As_l[buf][row][c4] = lv;
        }
    };

    v8f c[2][2];
#pragma unroll
    for (int mi = 0; mi < 2; ++mi)
#pragma unroll
        for (int ni = 0; ni < 2; ++ni) c[mi][ni] = (v8f)(0.0f);

    // prologue: stage tile 0 into buffer 0
    {
        float4 areg[4];
        issueB(0, 0);
        loadA(0, areg);
        storeA(areg, 0);
    }

    int cur = 0;
    for (int kb = 0; kb < K; kb += 32) {
#if USE_ASYNC_LDS
        __builtin_amdgcn_s_wait_asynccnt(0);
#endif
        __syncthreads();   // current buffer fully staged for all waves

        const bool hasNext = (kb + 32) < K;
        float4 areg[4];
        if (hasNext) {
            issueB(kb + 32, cur ^ 1);   // async: overlaps WMMAs below
            loadA(kb + 32, areg);       // global f32: waits land after WMMAs
        }

        // ---- fragments from current LDS buffer ----
        v16bf ah[2], al[2], bh[2], bl[2];
#pragma unroll
        for (int mi = 0; mi < 2; ++mi) {
            const int m = waveM * 32 + mi * 16 + lane16;
            ah[mi] = load_frag(&As_h[cur][m][0], k0);
            al[mi] = load_frag(&As_l[cur][m][0], k0);
        }
#pragma unroll
        for (int ni = 0; ni < 2; ++ni) {
            const int n = waveN * 32 + ni * 16 + lane16;
            bh[ni] = load_frag(&Bs_h[cur][n][0], k0);
            bl[ni] = load_frag(&Bs_l[cur][n][0], k0);
        }

        // ---- split-bf16 fp32 emulation: ah*bh + ah*bl + al*bh ----
#pragma unroll
        for (int mi = 0; mi < 2; ++mi)
#pragma unroll
            for (int ni = 0; ni < 2; ++ni) {
                c[mi][ni] = __builtin_amdgcn_wmma_f32_16x16x32_bf16(
                    false, ah[mi], false, bh[ni], (short)0, c[mi][ni], false, false);
                c[mi][ni] = __builtin_amdgcn_wmma_f32_16x16x32_bf16(
                    false, ah[mi], false, bl[ni], (short)0, c[mi][ni], false, false);
                c[mi][ni] = __builtin_amdgcn_wmma_f32_16x16x32_bf16(
                    false, al[mi], false, bh[ni], (short)0, c[mi][ni], false, false);
            }

        if (hasNext) storeA(areg, cur ^ 1);  // convert + LDS store after WMMAs
        cur ^= 1;
    }

    // ---- epilogue: bias + activation + store ----
#pragma unroll
    for (int mi = 0; mi < 2; ++mi)
#pragma unroll
        for (int ni = 0; ni < 2; ++ni) {
            int row0 = rowBase + waveM * 32 + mi * 16 + laneHi * 8;
            int col  = colBase + waveN * 32 + ni * 16 + lane16;
            if (col < N) {
                float bv = bias[col];
#pragma unroll
                for (int r = 0; r < 8; ++r) {
                    float v = c[mi][ni][r] + bv;
                    if (ACT == 1) v = v / (1.0f + __expf(-v));  // SiLU
                    out[(size_t)(row0 + r) * N + col] = v;
                }
            }
        }
}

// ---------------------------------------------------------------------------
// Row LayerNorm over H=1024, one 256-thread block per row
// ---------------------------------------------------------------------------
__global__ __launch_bounds__(256)
void ln_kernel(const float* __restrict__ h, const float* __restrict__ g,
               const float* __restrict__ be, float* __restrict__ out) {
    const int H = H_DIM;
    const int row = blockIdx.x, tid = threadIdx.x;
    __shared__ float red[256];
    const float* hr = h + (size_t)row * H;

    float4 x = *(const float4*)(hr + tid * 4);
    red[tid] = x.x + x.y + x.z + x.w;
    __syncthreads();
    for (int off = 128; off > 0; off >>= 1) {
        if (tid < off) red[tid] += red[tid + off];
        __syncthreads();
    }
    float mu = red[0] * (1.0f / H);
    __syncthreads();

    float d0 = x.x - mu, d1 = x.y - mu, d2 = x.z - mu, d3 = x.w - mu;
    red[tid] = d0 * d0 + d1 * d1 + d2 * d2 + d3 * d3;
    __syncthreads();
    for (int off = 128; off > 0; off >>= 1) {
        if (tid < off) red[tid] += red[tid + off];
        __syncthreads();
    }
    float inv = rsqrtf(red[0] * (1.0f / H) + 1e-5f);

    float4 gv = *(const float4*)(g + tid * 4);
    float4 bv = *(const float4*)(be + tid * 4);
    float4 o;
    o.x = d0 * inv * gv.x + bv.x;
    o.y = d1 * inv * gv.y + bv.y;
    o.z = d2 * inv * gv.z + bv.z;
    o.w = d3 * inv * gv.w + bv.w;
    *(float4*)(out + (size_t)row * H + tid * 4) = o;
}

// ---------------------------------------------------------------------------
// Evidential head: one wave32 per row, 8 rows per block
// ---------------------------------------------------------------------------
__device__ __forceinline__ float softplusf(float x) {
    return (x > 20.0f) ? x : log1pf(expf(x));
}

__device__ __forceinline__ float digammaf(float x) {
    float r = 0.0f;
    while (x < 6.0f) { r -= 1.0f / x; x += 1.0f; }
    float inv = 1.0f / x, inv2 = inv * inv;
    r += logf(x) - 0.5f * inv
       - inv2 * (1.0f / 12.0f - inv2 * (1.0f / 120.0f - inv2 * (1.0f / 252.0f)));
    return r;
}

__global__ __launch_bounds__(256)
void head_kernel(const float* __restrict__ logits,
                 float* __restrict__ alpha_o, float* __restrict__ ev_o,
                 float* __restrict__ ep_o, float* __restrict__ vac_o,
                 float* __restrict__ dis_o, float* __restrict__ ale_o) {
    const int K = K_CLS;
    const int tid = threadIdx.x, lane = tid & 31, w = tid >> 5;
    const int row = blockIdx.x * 8 + w;

    __shared__ float evs[8][104];
    __shared__ float bel[8][104];

    float evl[4];
#pragma unroll
    for (int t = 0; t < 4; ++t) {
        int j = lane + t * 32;
        evl[t] = 0.0f;
        if (j < K) {
            float e = softplusf(logits[(size_t)row * K + j]);
            evs[w][j] = e;
            evl[t] = e;
            alpha_o[(size_t)row * K + j] = e + 1.0f;
            ev_o[(size_t)row * K + j]   = e;
        }
    }
    __syncthreads();

    float S = (float)K;                        // sum(alpha) = K + sum(ev)
    for (int j = 0; j < K; ++j) S += evs[w][j];
    float invS = 1.0f / S;

#pragma unroll
    for (int t = 0; t < 4; ++t) {
        int j = lane + t * 32;
        if (j < K) {
            bel[w][j] = evl[t] * invS;
            ep_o[(size_t)row * K + j] = (evl[t] + 1.0f) * invS;
        }
    }
    if (lane == 0) vac_o[row] = (float)K * invS;
    __syncthreads();

    float bj[4];
#pragma unroll
    for (int t = 0; t < 4; ++t) {
        int j = lane + t * 32;
        bj[t] = (j < K) ? bel[w][j] : 0.0f;
    }
    float full = 0.0f, bsq = 0.0f;
    for (int i = 0; i < K; ++i) {
        float bi = bel[w][i];
#pragma unroll
        for (int t = 0; t < 4; ++t) {
            float b = bj[t];
            float s = bi + b + 1e-8f;
            full += (1.0f - fabsf(bi - b) / s) * bi * b;
        }
    }
#pragma unroll
    for (int t = 0; t < 4; ++t) bsq += bj[t] * bj[t];

    float dgS = digammaf(S);
    float ale = 0.0f;
#pragma unroll
    for (int t = 0; t < 4; ++t) {
        int j = lane + t * 32;
        if (j < K) {
            float a = evl[t] + 1.0f;
            ale += a * invS * (digammaf(a) - dgS);
        }
    }

#pragma unroll
    for (int off = 16; off > 0; off >>= 1) {
        full += __shfl_down(full, off, 32);
        bsq  += __shfl_down(bsq,  off, 32);
        ale  += __shfl_down(ale,  off, 32);
    }
    if (lane == 0) {
        dis_o[row] = 0.5f * (full - bsq);
        ale_o[row] = -ale;
    }
}

// ---------------------------------------------------------------------------
extern "C" void kernel_launch(void* const* d_in, const int* in_sizes, int n_in,
                              void* d_out, int out_size, void* d_ws, size_t ws_size,
                              hipStream_t stream) {
    (void)in_sizes; (void)n_in; (void)out_size; (void)ws_size;
    const float* x    = (const float*)d_in[0];
    const float* W1   = (const float*)d_in[1];
    const float* b1   = (const float*)d_in[2];
    const float* g    = (const float*)d_in[3];
    const float* beta = (const float*)d_in[4];
    const float* W2   = (const float*)d_in[5];
    const float* b2   = (const float*)d_in[6];
    const float* W3   = (const float*)d_in[7];
    const float* b3   = (const float*)d_in[8];

    const int Bn = B_ROWS, Dd = D_DIM, Hh = H_DIM, Kk = K_CLS;

    float* ws0 = (float*)d_ws;                     // h1 / h2 (B x H)
    float* ws1 = ws0 + (size_t)Bn * Hh;            // layernorm out (B x H)
    float* lgt = ws1 + (size_t)Bn * Hh;            // logits (B x K)
    __bf16* bfb   = (__bf16*)(lgt + (size_t)Bn * Kk);
    __bf16* W1t_h = bfb;                           // [H x D]
    __bf16* W1t_l = W1t_h + (size_t)Hh * Dd;
    __bf16* W2t_h = W1t_l + (size_t)Hh * Dd;       // [H x H]
    __bf16* W2t_l = W2t_h + (size_t)Hh * Hh;
    __bf16* W3t_h = W2t_l + (size_t)Hh * Hh;       // [K x H]
    __bf16* W3t_l = W3t_h + (size_t)Kk * Hh;

    float* out     = (float*)d_out;
    float* alpha_o = out;
    float* ev_o    = out + (size_t)Bn * Kk;
    float* ep_o    = out + 2 * (size_t)Bn * Kk;
    float* vac_o   = out + 3 * (size_t)Bn * Kk;
    float* dis_o   = vac_o + Bn;
    float* ale_o   = dis_o + Bn;

    dim3 blk(256);
    // one-time weight prep: split + transpose to bf16 hi/lo [N x K]
    transpose_split<<<dim3(32, 32), blk, 0, stream>>>(W1, W1t_h, W1t_l, Dd, Hh);
    transpose_split<<<dim3(32, 32), blk, 0, stream>>>(W2, W2t_h, W2t_l, Hh, Hh);
    transpose_split<<<dim3(4, 32),  blk, 0, stream>>>(W3, W3t_h, W3t_l, Hh, Kk);

    // h1 = silu(x @ W1 + b1)
    gemm_bias_act<1><<<dim3(Bn / 128, Hh / 64), blk, 0, stream>>>(
        x, W1t_h, W1t_l, b1, ws0, Bn, Hh, Dd);
    // h1 = LN(h1)
    ln_kernel<<<dim3(Bn), blk, 0, stream>>>(ws0, g, beta, ws1);
    // h2 = silu(h1 @ W2 + b2)
    gemm_bias_act<1><<<dim3(Bn / 128, Hh / 64), blk, 0, stream>>>(
        ws1, W2t_h, W2t_l, b2, ws0, Bn, Hh, Hh);
    // logits = h2 @ W3 + b3
    gemm_bias_act<0><<<dim3(Bn / 128, (Kk + 63) / 64), blk, 0, stream>>>(
        ws0, W3t_h, W3t_l, b3, lgt, Bn, Kk, Hh);
    // evidential statistics
    head_kernel<<<dim3(Bn / 8), blk, 0, stream>>>(
        lgt, alpha_o, ev_o, ep_o, vac_o, dis_o, ale_o);
}
